// PKA2_Net_36266703848086
// MI455X (gfx1250) — matrix operand order repaired and checked by reading, not compiled, not run on hardware
//
#include <hip/hip_runtime.h>

// ---------------------------------------------------------------------------
// PKA2-Net forward on gfx1250 (MI455X). All convolutions run as implicit-GEMM
// on V_WMMA_F32_16X16X32_F16 (wave32). Block tile 64x128x32, 8 waves in a
// 2(M)x4(N) grid, 4 WMMAs per wave per K-step. LDS tiles are stored in
// fragment order so fragment loads are ds_load_b128 pairs. Weights are
// pre-converted to f16 per conv; the A tile is then staged to LDS with
// GLOBAL_LOAD_ASYNC_TO_LDS_B128 (one per lane) when available, fenced with
// s_wait_asynccnt. Activations are f16 end-to-end; f32 accumulate;
// residual-add / ReLU fused in the epilogue.
// ---------------------------------------------------------------------------

typedef _Float16 half_t;
typedef __attribute__((ext_vector_type(16))) _Float16 v16h;
typedef __attribute__((ext_vector_type(8)))  _Float16 v8h;
typedef __attribute__((ext_vector_type(8)))  float    v8f;
typedef int v4i32 __attribute__((__vector_size__(16)));   // matches builtin param

#define CB_M 64
#define CB_N 128
#define CB_K 32
#define CB_THREADS 256
#define FR_PITCH 24   // halves per lane row (48B: 16B-aligned, conflict-spread)
#define T_T 81        // number of SEBS templates (9x9 grid)

#if defined(__has_builtin)
#if __has_builtin(__builtin_amdgcn_global_load_async_to_lds_b128)
#define HAVE_ASYNC_LDS 1
#endif
#if __has_builtin(__builtin_amdgcn_s_wait_asynccnt)
#define HAVE_WAIT_ASYNC_BUILTIN 1
#endif
#endif

static __device__ __forceinline__ void wait_async0() {
#if defined(HAVE_ASYNC_LDS)
#if defined(HAVE_WAIT_ASYNC_BUILTIN)
  __builtin_amdgcn_s_wait_asynccnt(0);
#else
  asm volatile("s_wait_asynccnt 0x0" ::: "memory");
#endif
#endif
}

static inline int cdiv(int a, int b) { return (a + b - 1) / b; }

// -------------------------------- kernels ----------------------------------

__global__ void k_f32_to_f16(const float* __restrict__ x, half_t* __restrict__ y, int n) {
  int i = blockIdx.x * blockDim.x + threadIdx.x;
  if (i < n) y[i] = (half_t)x[i];
}

// Implicit-GEMM convolution: D[M=Cout, N=B*Ho*Wo] = W[M,K] * im2col(X)[K,N]
// w is pre-converted f16, row-major M x K.
__global__ __launch_bounds__(CB_THREADS)
void k_conv_wmma(const half_t* __restrict__ x, const half_t* __restrict__ w,
                 half_t* __restrict__ y, const half_t* __restrict__ res,
                 int B, int Cin, int H, int W, int Cout, int KH, int KW,
                 int stride, int pad, int Ho, int Wo, int relu)
{
  const int tid  = threadIdx.x;
  const int lane = tid & 31;
  const int wv   = tid >> 5;          // 0..7
  const int wm   = wv >> 2;           // 0..1  (M sub-tile)
  const int wn   = wv & 3;            // 0..3  (N sub-tile)
  const int M = Cout;
  const int N = B * Ho * Wo;
  const int K = Cin * KH * KW;
  const int KHW = KH * KW;
  const int m0 = blockIdx.y * CB_M;
  const int n0 = blockIdx.x * CB_N;

  // Fragment-ordered LDS: [frag][lane][16 halves contiguous, 16B-aligned]
  __shared__ __align__(16) half_t AsF[4][32][FR_PITCH];   // 64x32 A tile
  __shared__ __align__(16) half_t BsF[8][32][FR_PITCH];   // 32x128 B tile

  v8f zero = {};
  v8f acc[2][2];
#pragma unroll
  for (int i = 0; i < 2; ++i)
#pragma unroll
    for (int j = 0; j < 2; ++j) acc[i][j] = zero;

  // ---- per-thread staging precompute ----
  // Async A path: each of the 256 threads moves one 16B chunk (8 K-halves).
  const bool asyncA = ((K & 31) == 0) && ((M & 63) == 0);
  const int  mmA2 = tid >> 2;         // 0..63 : A row
  const int  gA   = tid & 3;          // chunk: kk = 8*gA .. 8*gA+7
  half_t* a_dst = &AsF[mmA2 >> 4][(mmA2 & 15) + ((gA & 1) ? 16 : 0)][(gA & 2) ? 8 : 0];
  // Fallback scalar A path: kk fixed per thread (tid&31), mm sweeps.
  const int kkA   = tid & 31;
  const int slotA = (kkA & 7) + ((kkA & 16) ? 8 : 0);   // inverse of A lane map
  const int loffA = (kkA & 8) ? 16 : 0;
  // B staging: nn fixed per thread (tid&127), kk sweeps (tid>>7) + 2*t.
  const int nnB   = tid & 127;
  const int fragB = nnB >> 4;
  const int lnB   = nnB & 15;
  const int gn    = n0 + nnB;
  const bool vn   = gn < N;
  int ih0 = 0, iw0 = 0;
  size_t xbase = 0;
  if (vn) {
    int b  = gn / (Ho * Wo);
    int pp = gn - b * (Ho * Wo);
    int ho = pp / Wo, wo = pp - ho * Wo;
    ih0 = ho * stride - pad;
    iw0 = wo * stride - pad;
    xbase = (size_t)b * Cin * H * W;
  }

  for (int k0 = 0; k0 < K; k0 += CB_K) {
    // speculative prefetch of next weight K-slab
    if (k0 + CB_K < K)
      __builtin_prefetch(w + (size_t)(m0 + (tid & 63)) * K + k0 + CB_K, 0, 1);

    // ---- stage A (64x32), fragment order ----
#if defined(HAVE_ASYNC_LDS)
    if (asyncA) {
      // One async 16B global->LDS transfer per lane; both sides 16B-aligned.
      const half_t* src = w + (size_t)(m0 + mmA2) * K + k0 + 8 * gA;
      __builtin_amdgcn_global_load_async_to_lds_b128(
          (__attribute__((address_space(1))) v4i32*)src,
          (__attribute__((address_space(3))) v4i32*)a_dst,
          0, 0);
    } else
#endif
    {
      const int gk = k0 + kkA;
      const bool vk = gk < K;
#pragma unroll
      for (int t = 0; t < 8; ++t) {
        int mm = (tid >> 5) + 8 * t;
        int gm = m0 + mm;
        AsF[mm >> 4][(mm & 15) + loffA][slotA] =
            (vk && gm < M) ? w[(size_t)gm * K + gk] : (half_t)0.f;
      }
    }
    // ---- stage B (32x128) im2col gather, fragment order ----
#pragma unroll
    for (int t = 0; t < 16; ++t) {
      int kk = (tid >> 7) + 2 * t;
      int gk = k0 + kk;
      half_t v = (half_t)0.f;
      if (vn && gk < K) {
        int ci = gk / KHW;
        int rr = gk - ci * KHW;
        int kh = rr / KW, kw = rr - kh * KW;
        int ih = ih0 + kh, iw = iw0 + kw;
        if (ih >= 0 && ih < H && iw >= 0 && iw < W)
          v = x[xbase + ((size_t)ci * H + ih) * W + iw];
      }
      BsF[fragB][lnB + ((kk & 8) ? 16 : 0)][(kk & 7) + ((kk & 16) ? 8 : 0)] = v;
    }
#if defined(HAVE_ASYNC_LDS)
    if (asyncA) wait_async0();
#endif
    __syncthreads();

    // ---- fragment loads (b128 pairs) + 4 WMMAs ----
    v8h a0l = *(const v8h*)&AsF[2 * wm + 0][lane][0];
    v8h a0h = *(const v8h*)&AsF[2 * wm + 0][lane][8];
    v8h a1l = *(const v8h*)&AsF[2 * wm + 1][lane][0];
    v8h a1h = *(const v8h*)&AsF[2 * wm + 1][lane][8];
    v8h b0l = *(const v8h*)&BsF[2 * wn + 0][lane][0];
    v8h b0h = *(const v8h*)&BsF[2 * wn + 0][lane][8];
    v8h b1l = *(const v8h*)&BsF[2 * wn + 1][lane][0];
    v8h b1h = *(const v8h*)&BsF[2 * wn + 1][lane][8];
    v16h a0 = __builtin_shufflevector(a0l, a0h, 0,1,2,3,4,5,6,7,8,9,10,11,12,13,14,15);
    v16h a1 = __builtin_shufflevector(a1l, a1h, 0,1,2,3,4,5,6,7,8,9,10,11,12,13,14,15);
    v16h b0 = __builtin_shufflevector(b0l, b0h, 0,1,2,3,4,5,6,7,8,9,10,11,12,13,14,15);
    v16h b1 = __builtin_shufflevector(b1l, b1h, 0,1,2,3,4,5,6,7,8,9,10,11,12,13,14,15);

    acc[0][0] = __builtin_amdgcn_wmma_f32_16x16x32_f16(false, a0, false, b0,
                                                       (short)0, acc[0][0], false, false);
    acc[0][1] = __builtin_amdgcn_wmma_f32_16x16x32_f16(false, a0, false, b1,
                                                       (short)0, acc[0][1], false, false);
    acc[1][0] = __builtin_amdgcn_wmma_f32_16x16x32_f16(false, a1, false, b0,
                                                       (short)0, acc[1][0], false, false);
    acc[1][1] = __builtin_amdgcn_wmma_f32_16x16x32_f16(false, a1, false, b1,
                                                       (short)0, acc[1][1], false, false);
    __syncthreads();
  }

  // ---- epilogue: C/D layout — VGPR r: lanes 0-15 -> M=r, lanes 16-31 -> M=r+8
#pragma unroll
  for (int i = 0; i < 2; ++i)
#pragma unroll
    for (int j = 0; j < 2; ++j) {
      int mb = m0 + wm * 32 + i * 16;
      int nb = n0 + wn * 32 + j * 16;
#pragma unroll
      for (int r = 0; r < 8; ++r) {
        int mm = mb + r + ((lane >> 4) << 3);
        int nn = nb + (lane & 15);
        if (mm < M && nn < N) {
          int b  = nn / (Ho * Wo);
          int pp = nn - b * (Ho * Wo);
          size_t o = (size_t)(b * Cout + mm) * (Ho * Wo) + pp;
          float v = acc[i][j][r];
          if (res)  v += (float)res[o];
          if (relu) v = v > 0.f ? v : 0.f;
          y[o] = (half_t)v;
        }
      }
    }
}

// Per-channel batch statistics (training-mode BN, biased variance).
__global__ void k_bn_stats(const half_t* __restrict__ x, float* __restrict__ mean,
                           float* __restrict__ rstd, int B, int C, int HW, float eps)
{
  int c = blockIdx.x;
  __shared__ float s1[256], s2[256];
  float a = 0.f, q = 0.f;
  for (int i = threadIdx.x; i < B * HW; i += blockDim.x) {
    int bb = i / HW, p = i - bb * HW;
    float v = (float)x[((size_t)(bb * C + c)) * HW + p];
    a += v; q += v * v;
  }
  s1[threadIdx.x] = a; s2[threadIdx.x] = q;
  __syncthreads();
  for (int s = blockDim.x / 2; s > 0; s >>= 1) {
    if (threadIdx.x < s) { s1[threadIdx.x] += s1[threadIdx.x + s]; s2[threadIdx.x] += s2[threadIdx.x + s]; }
    __syncthreads();
  }
  if (threadIdx.x == 0) {
    float n  = (float)(B * HW);
    float mu = s1[0] / n;
    float va = s2[0] / n - mu * mu;
    mean[c] = mu;
    rstd[c] = rsqrtf(va + eps);
  }
}

__global__ void k_bn_apply(half_t* __restrict__ x, const float* __restrict__ g,
                           const float* __restrict__ b, const float* __restrict__ mean,
                           const float* __restrict__ rstd, int C, int HW,
                           long long n, int relu)
{
  long long i = (long long)blockIdx.x * blockDim.x + threadIdx.x;
  if (i >= n) return;
  int c = (int)((i / HW) % C);
  float v = (float)x[i];
  v = g[c] * (v - mean[c]) * rstd[c] + b[c];
  if (relu) v = v > 0.f ? v : 0.f;
  x[i] = (half_t)v;
}

__global__ void k_maxpool3s2(const half_t* __restrict__ x, half_t* __restrict__ y,
                             int B, int C, int H, int W, int Ho, int Wo)
{
  long long n = (long long)B * C * Ho * Wo;
  long long i = (long long)blockIdx.x * blockDim.x + threadIdx.x;
  if (i >= n) return;
  int wo = (int)(i % Wo); long long t = i / Wo;
  int ho = (int)(t % Ho); t /= Ho;
  int c  = (int)(t % C);  int b = (int)(t / C);
  float m = -3.4e38f;
  for (int dh = 0; dh < 3; ++dh)
    for (int dw = 0; dw < 3; ++dw) {
      int ih = ho * 2 - 1 + dh, iw = wo * 2 - 1 + dw;
      if (ih >= 0 && ih < H && iw >= 0 && iw < W) {
        float v = (float)x[((size_t)(b * C + c) * H + ih) * W + iw];
        m = v > m ? v : m;
      }
    }
  y[i] = (half_t)m;
}

// count[t] = #{(x,y) : tem(x,y,t) > 0}; tem is the raw (T,S,S) buffer read with
// (S,S,T) strides (raw reshape), i.e. tem_flat[p*T + t].
__global__ void k_sebs_count(const half_t* __restrict__ tem, float* __restrict__ cnt, int SS)
{
  int t = blockIdx.x;
  __shared__ float s[256];
  float a = 0.f;
  for (int p = threadIdx.x; p < SS; p += blockDim.x)
    a += ((float)tem[(size_t)p * T_T + t] > 0.f) ? 1.f : 0.f;
  s[threadIdx.x] = a;
  __syncthreads();
  for (int k = blockDim.x / 2; k > 0; k >>= 1) {
    if (threadIdx.x < k) s[threadIdx.x] += s[threadIdx.x + k];
    __syncthreads();
  }
  if (threadIdx.x == 0) cnt[t] = s[0];
}

// gap[b,t,c] = (sum_p tem[p,t] * x[b,c,p]) / SS / count[t]
__global__ void k_sebs_gap(const half_t* __restrict__ tem, const half_t* __restrict__ x,
                           const float* __restrict__ cnt, float* __restrict__ gap,
                           int C, int SS)
{
  int t = blockIdx.x, b = blockIdx.y;
  int c = threadIdx.x;
  if (c >= C) return;
  const half_t* xc = x + ((size_t)(b * C + c)) * SS;
  float a = 0.f;
  for (int p = 0; p < SS; ++p)
    a += (float)tem[(size_t)p * T_T + t] * (float)xc[p];
  gap[((size_t)b * T_T + t) * C + c] = a / (float)SS / cnt[t];
}

// Iterative top-3 over templates: exact max / (>=m) mask (ties included) / zero.
__global__ void k_sebs_topk(const float* __restrict__ gap, half_t* __restrict__ atten,
                            int Bn, int C)
{
  int i = blockIdx.x * blockDim.x + threadIdx.x;
  if (i >= Bn * C) return;
  int b = i / C, c = i - b * C;
  float g[T_T];
  for (int t = 0; t < T_T; ++t) g[t] = gap[((size_t)b * T_T + t) * C + c];
  for (int k = 0; k < 3; ++k) {
    float m = -3.4e38f;
    for (int t = 0; t < T_T; ++t) m = g[t] > m ? g[t] : m;
    for (int t = 0; t < T_T; ++t) {
      bool hit = g[t] >= m;
      atten[(((size_t)b * T_T + t) * 3 + k) * C + c] = hit ? (half_t)1.f : (half_t)0.f;
      if (hit) g[t] = 0.f;
    }
  }
}

// y[b,:,p] = concat( x[b,:,p],  x[b,c,p] * sum_t tem[p,t]*atten[b,t,k,c] )
__global__ void k_sebs_out(const half_t* __restrict__ tem, const half_t* __restrict__ x,
                           const half_t* __restrict__ atten, half_t* __restrict__ y,
                           int C, int S)
{
  int p = blockIdx.x;           // pixel index in [0, S*S)
  int b = blockIdx.y;
  int SS = S * S;
  __shared__ float tm[T_T];
  for (int t = threadIdx.x; t < T_T; t += blockDim.x)
    tm[t] = (float)tem[(size_t)p * T_T + t];
  __syncthreads();
  int C4 = 4 * C;
  for (int ch = threadIdx.x; ch < C4; ch += blockDim.x) {
    float o;
    if (ch < C) {
      o = (float)x[((size_t)(b * C + ch)) * SS + p];
    } else {
      int k = (ch - C) / C;
      int c = (ch - C) - k * C;
      const half_t* at = atten + ((size_t)b * T_T * 3 + k) * C + c;  // +t*3*C per t
      float wsum = 0.f;
      for (int t = 0; t < T_T; ++t)
        wsum += tm[t] * (float)at[(size_t)t * 3 * C];
      o = (float)x[((size_t)(b * C + c)) * SS + p] * wsum;
    }
    y[((size_t)(b * C4 + ch)) * SS + p] = (half_t)o;
  }
}

__global__ void k_avgpool(const half_t* __restrict__ x, float* __restrict__ y,
                          int BC, int HW)
{
  int i = blockIdx.x * blockDim.x + threadIdx.x;
  if (i >= BC) return;
  const half_t* p = x + (size_t)i * HW;
  float a = 0.f;
  for (int j = 0; j < HW; ++j) a += (float)p[j];
  y[i] = a / (float)HW;
}

__global__ void k_fc(const float* __restrict__ feat, const float* __restrict__ w,
                     const float* __restrict__ bias, float* __restrict__ out,
                     int Cin, int Nout, int Btot)
{
  int i = blockIdx.x * blockDim.x + threadIdx.x;
  if (i >= Btot * Nout) return;
  int b = i / Nout, n = i - b * Nout;
  float a = bias[n];
  for (int c = 0; c < Cin; ++c) a += feat[b * Cin + c] * w[(size_t)n * Cin + c];
  out[i] = a;
}

// ------------------------------ host helpers -------------------------------

struct RB { const float *w1, *g1, *b1, *w2, *w3; };

static inline void swp(half_t*& a, half_t*& b) { half_t* t = a; a = b; b = t; }

static void run_conv(hipStream_t st, const half_t* x, const float* w, half_t* wc,
                     half_t* y, const half_t* res, int B, int Cin, int H, int W,
                     int Cout, int KH, int KW, int stride, int pad, int relu)
{
  int K = Cin * KH * KW;
  k_f32_to_f16<<<cdiv(Cout * K, 256), 256, 0, st>>>(w, wc, Cout * K);
  int Ho = (H + 2 * pad - KH) / stride + 1;
  int Wo = (W + 2 * pad - KW) / stride + 1;
  dim3 grid(cdiv(B * Ho * Wo, CB_N), cdiv(Cout, CB_M));
  k_conv_wmma<<<grid, CB_THREADS, 0, st>>>(x, wc, y, res, B, Cin, H, W,
                                           Cout, KH, KW, stride, pad, Ho, Wo, relu);
}

static void run_bn(hipStream_t st, half_t* x, const float* g, const float* b,
                   float* mean, float* rstd, int B, int C, int HW, int relu)
{
  k_bn_stats<<<C, 256, 0, st>>>(x, mean, rstd, B, C, HW, 1e-5f);
  long long n = (long long)B * C * HW;
  k_bn_apply<<<cdiv((int)n, 256), 256, 0, st>>>(x, g, b, mean, rstd, C, HW, n, relu);
}

static void resblock(hipStream_t st, half_t*& cur, half_t*& t1, half_t*& t2, half_t* wc,
                     const RB& p, int B, int C, int S, float* mean, float* rstd)
{
  run_conv(st, cur, p.w1, wc, t1, nullptr, B, C, S, S, C, 3, 3, 1, 1, 0);
  run_bn(st, t1, p.g1, p.b1, mean, rstd, B, C, S * S, 0);
  run_conv(st, t1, p.w2, wc, t2, nullptr, B, C, S, S, 4 * C, 1, 1, 1, 0, 1);
  run_conv(st, t2, p.w3, wc, t1, cur, B, 4 * C, S, S, C, 1, 1, 1, 0, 0);  // +residual
  swp(cur, t1);
}

static void resdown(hipStream_t st, half_t*& cur, half_t*& t1, half_t*& t2, half_t* wc,
                    const RB& p, int B, int Cin, int Cout, int& S,
                    float* mean, float* rstd)
{
  run_conv(st, cur, p.w1, wc, t1, nullptr, B, Cin, S, S, Cout, 3, 3, 2, 1, 0);
  S = (S + 2 - 3) / 2 + 1;
  run_bn(st, t1, p.g1, p.b1, mean, rstd, B, Cout, S * S, 0);
  run_conv(st, t1, p.w2, wc, t2, nullptr, B, Cout, S, S, 4 * Cout, 1, 1, 1, 0, 1);
  run_conv(st, t2, p.w3, wc, t1, nullptr, B, 4 * Cout, S, S, Cout, 1, 1, 1, 0, 0);
  swp(cur, t1);
}

static void sebs(hipStream_t st, half_t*& cur, half_t*& t1, const half_t* tem,
                 int B, int C, int S, float* cnt, float* gap, half_t* atten)
{
  int SS = S * S;
  k_sebs_count<<<T_T, 256, 0, st>>>(tem, cnt, SS);
  k_sebs_gap<<<dim3(T_T, B), C, 0, st>>>(tem, cur, cnt, gap, C, SS);
  k_sebs_topk<<<cdiv(B * C, 128), 128, 0, st>>>(gap, atten, B, C);
  k_sebs_out<<<dim3(SS, B), 128, 0, st>>>(tem, cur, atten, t1, C, S);
  swp(cur, t1);   // cur now has 4C channels
}

// ------------------------------ kernel_launch ------------------------------

extern "C" void kernel_launch(void* const* d_in, const int* in_sizes, int n_in,
                              void* d_out, int out_size, void* d_ws, size_t ws_size,
                              hipStream_t stream)
{
  (void)in_sizes; (void)n_in; (void)out_size; (void)ws_size;

  // --- consume inputs in setup_inputs() flattening order ---
  int idx = 0;
  auto F = [&]() -> const float* { return (const float*)d_in[idx++]; };

  const float* X       = F();                                   // (2,3,512,512)
  const float* w_conv1 = F();                                   // (64,3,7,7)
  const float* g11     = F();
  const float* b11     = F();
  RB blk1[3]; for (int k = 0; k < 3; ++k) blk1[k] = { F(), F(), F(), F(), F() };
  const float* w1x1_1  = F();                                   // (64,256,1,1)
  RB down2 = { F(), F(), F(), F(), F() };
  RB blk2[3]; for (int k = 0; k < 3; ++k) blk2[k] = { F(), F(), F(), F(), F() };
  const float* w1x1_2  = F();                                   // (128,512,1,1)
  RB down3 = { F(), F(), F(), F(), F() };
  RB blk3[5]; for (int k = 0; k < 5; ++k) blk3[k] = { F(), F(), F(), F(), F() };
  const float* w1x1_3  = F();                                   // (256,1024,1,1)
  RB down4 = { F(), F(), F(), F(), F() };
  RB blk4[2]; for (int k = 0; k < 2; ++k) blk4[k] = { F(), F(), F(), F(), F() };
  const float* fcw = F();                                       // (1000,512)
  const float* fcb = F();                                       // (1000)
  const half_t* t56 = (const half_t*)d_in[idx++];               // (81,128,128) f16
  const half_t* t28 = (const half_t*)d_in[idx++];               // (81,64,64)  f16
  const half_t* t14 = (const half_t*)d_in[idx++];               // (81,32,32)  f16

  // --- workspace carve (peak act = 2*64*256*256 = 8,388,608 f16 elements) ---
  const size_t ACT_BYTES = (size_t)8388608 * sizeof(half_t);
  char* wsb = (char*)d_ws;
  half_t* b0 = (half_t*)(wsb + 0 * ACT_BYTES);
  half_t* b1 = (half_t*)(wsb + 1 * ACT_BYTES);
  half_t* b2 = (half_t*)(wsb + 2 * ACT_BYTES);
  char* sm = wsb + 3 * ACT_BYTES;
  half_t* wcvt  = (half_t*)sm; sm += (size_t)2359296 * sizeof(half_t); // largest weight (512x4608)
  float*  mean  = (float*)sm;  sm += 2048 * sizeof(float);
  float*  rstd  = (float*)sm;  sm += 2048 * sizeof(float);
  float*  cnt   = (float*)sm;  sm += 128 * sizeof(float);
  float*  gap   = (float*)sm;  sm += (size_t)2 * T_T * 256 * sizeof(float);
  half_t* atten = (half_t*)sm; sm += (size_t)2 * T_T * 3 * 256 * sizeof(half_t);
  float*  feat  = (float*)sm;  sm += 2 * 512 * sizeof(float);

  const int B = 2;

  // --- stem ---
  int n_x = B * 3 * 512 * 512;
  k_f32_to_f16<<<cdiv(n_x, 256), 256, 0, stream>>>(X, b0, n_x);
  run_conv(stream, b0, w_conv1, wcvt, b1, nullptr, B, 3, 512, 512, 64, 7, 7, 2, 3, 0);
  run_bn(stream, b1, g11, b11, mean, rstd, B, 64, 256 * 256, 1);
  {
    long long n = (long long)B * 64 * 128 * 128;
    k_maxpool3s2<<<cdiv((int)n, 256), 256, 0, stream>>>(b1, b0, B, 64, 256, 256, 128, 128);
  }

  half_t* cur = b0; half_t* u1 = b1; half_t* u2 = b2;
  int C = 64, S = 128;

  // --- stage 1 ---
  for (int k = 0; k < 3; ++k) resblock(stream, cur, u1, u2, wcvt, blk1[k], B, C, S, mean, rstd);
  sebs(stream, cur, u1, t56, B, C, S, cnt, gap, atten);                  // -> 256 ch
  run_conv(stream, cur, w1x1_1, wcvt, u1, nullptr, B, 4 * C, S, S, 64, 1, 1, 1, 0, 0);
  swp(cur, u1);

  // --- stage 2 ---
  resdown(stream, cur, u1, u2, wcvt, down2, B, 64, 128, S, mean, rstd);  // S 128->64
  C = 128;
  for (int k = 0; k < 3; ++k) resblock(stream, cur, u1, u2, wcvt, blk2[k], B, C, S, mean, rstd);
  sebs(stream, cur, u1, t28, B, C, S, cnt, gap, atten);                  // -> 512 ch
  run_conv(stream, cur, w1x1_2, wcvt, u1, nullptr, B, 4 * C, S, S, 128, 1, 1, 1, 0, 0);
  swp(cur, u1);

  // --- stage 3 ---
  resdown(stream, cur, u1, u2, wcvt, down3, B, 128, 256, S, mean, rstd); // S 64->32
  C = 256;
  for (int k = 0; k < 5; ++k) resblock(stream, cur, u1, u2, wcvt, blk3[k], B, C, S, mean, rstd);
  sebs(stream, cur, u1, t14, B, C, S, cnt, gap, atten);                  // -> 1024 ch
  run_conv(stream, cur, w1x1_3, wcvt, u1, nullptr, B, 4 * C, S, S, 256, 1, 1, 1, 0, 0);
  swp(cur, u1);

  // --- stage 4 ---
  resdown(stream, cur, u1, u2, wcvt, down4, B, 256, 512, S, mean, rstd); // S 32->16
  C = 512;
  for (int k = 0; k < 2; ++k) resblock(stream, cur, u1, u2, wcvt, blk4[k], B, C, S, mean, rstd);

  // --- head ---
  k_avgpool<<<cdiv(B * 512, 128), 128, 0, stream>>>(cur, feat, B * 512, S * S);
  k_fc<<<cdiv(B * 1000, 128), 128, 0, stream>>>(feat, fcw, fcb, (float*)d_out, 512, 1000, B);
}
